// EdgeGATv2Conv_39599598469260
// MI455X (gfx1250) — matrix operand order repaired and checked by reading, not compiled, not run on hardware
//
#include <hip/hip_runtime.h>
#include <hip/hip_bf16.h>

#define N_NODES   50000
#define N_EDGES   1000000
#define D_IN      64
#define D_OUT     64
#define EDGE_DIM  16
#define NEG_SLOPE 0.2f

typedef __attribute__((ext_vector_type(16))) __bf16 v16bf;
typedef __attribute__((ext_vector_type(8)))  float  v8f;

// ---------------------------------------------------------------------------
// Kernel 0: out = bias (broadcast), mEnc = 0 (ordered-encoded floor),
// denom = 0. Harness poisons d_out/d_ws, so every call re-initializes.
// ---------------------------------------------------------------------------
__global__ void gat_init(float* __restrict__ out, const float* __restrict__ bias,
                         unsigned* __restrict__ mEnc, float* __restrict__ denom) {
  int i = blockIdx.x * blockDim.x + threadIdx.x;
  if (i < N_NODES * D_OUT) out[i] = bias[i & (D_OUT - 1)];
  if (i < N_NODES) { mEnc[i] = 0u; denom[i] = 0.0f; }
}

// ---------------------------------------------------------------------------
// Kernel 1: x_l = x@W_l + b_l, x_r = x@W_r + b_r via v_wmma_f32_16x16x32_bf16.
// Block = 256 threads = 8 waves; wave w: matrix (w>>2), output col-tile (w&3).
// Grid = 50000/16 = 3125 row tiles (exact); no divergence -> EXEC all 1s.
// ---------------------------------------------------------------------------
__global__ void node_transform_wmma(const float* __restrict__ x,
                                    const float* __restrict__ W_l,
                                    const float* __restrict__ b_l,
                                    const float* __restrict__ W_r,
                                    const float* __restrict__ b_r,
                                    float* __restrict__ x_l,
                                    float* __restrict__ x_r) {
  const int tid   = threadIdx.x;
  const int wave  = tid >> 5;
  const int lane  = tid & 31;
  const int row0  = blockIdx.x << 4;     // 16-node row tile
  const int which = wave >> 2;           // 0 -> (W_l,b_l,x_l), 1 -> (W_r,b_r,x_r)
  const int ncol0 = (wave & 3) << 4;     // 16-wide column tile of D_OUT

  const float* W   = which ? W_r : W_l;
  const float* bv  = which ? b_r : b_l;
  float*       dst = which ? x_r : x_l;

  const int  m  = lane & 15;
  const int  n  = lane & 15;
  const bool hi = (lane >= 16);
  const int  ka = hi ? 8  : 0;           // A K sub-base within 16-wide half
  const int  kb = hi ? 16 : 0;           // B K sub-base

  v8f acc = {};
  #pragma unroll
  for (int k0 = 0; k0 < D_IN; k0 += 32) {
    const float* xrow = x + (size_t)(row0 + m) * D_IN + k0;
    v16bf a, b;
    #pragma unroll
    for (int j = 0; j < 8; ++j) {
      a[j]     = (__bf16)xrow[ka + j];          // K = ka + j
      a[j + 8] = (__bf16)xrow[16 + ka + j];     // K = 16 + ka + j
    }
    #pragma unroll
    for (int j = 0; j < 16; ++j)
      b[j] = (__bf16)W[(size_t)(k0 + (j < 8 ? ka + j : 8 + ka + j)) * D_OUT + ncol0 + n];
    acc = __builtin_amdgcn_wmma_f32_16x16x32_bf16(
        /*neg_a=*/false, a, /*neg_b=*/false, b,
        /*c_mod=*/(short)0, acc, /*reuse_a=*/false, /*reuse_b=*/false);
  }

  const float bb = bv[ncol0 + n];
  const int   mb = hi ? 8 : 0;
  #pragma unroll
  for (int v = 0; v < 8; ++v)
    dst[(size_t)(row0 + mb + v) * D_OUT + ncol0 + n] = acc[v] + bb;
}

// ---------------------------------------------------------------------------
// Kernel 2: per-edge attention logit with WMMA edge transform.
//   E_tile[16 edges x 16 cols] = edge_attr_tile @ W_e   (K=16 zero-padded to 32)
//   logit[e] = sum_cols att[c] * leaky_relu(x_l[src][c] + x_r[dst][c] + E)
// Block = 256 thr = 8 waves = 2 edge groups x 4 col tiles = 32 edges/block.
// C-layout (lane = col n, 8 rows/lane) feeds the per-edge gather + LeakyReLU
// directly; row dot-product reduces via shfl_xor butterfly (16-lane halves)
// then ds_add_f32 across the 4 column-tile waves. 1M % 32 == 0 -> EXEC all 1s.
// ---------------------------------------------------------------------------
__global__ void edge_logits_wmma(const float* __restrict__ x_l,
                                 const float* __restrict__ x_r,
                                 const int* __restrict__ ei,
                                 const float* __restrict__ edge_attr,
                                 const float* __restrict__ W_e,
                                 const float* __restrict__ att,
                                 float* __restrict__ logits,
                                 unsigned* __restrict__ mEnc) {
  __shared__ float sLogit[32];

  const int tid   = threadIdx.x;
  const int wave  = tid >> 5;
  const int lane  = tid & 31;
  const int g     = wave >> 2;                 // edge sub-group (0/1)
  const int ncol0 = (wave & 3) << 4;           // 16-wide column tile
  const int e0    = blockIdx.x * 32 + g * 16;  // first edge of this wave's tile

  if (tid < 32) sLogit[tid] = 0.0f;
  __syncthreads();

  const int  mrow = lane & 15;
  const int  n    = lane & 15;
  const bool hi   = (lane >= 16);

  // A: 16 edges x 32 (bf16). Real K = 0..15 (EDGE_DIM), K = 16..31 zero.
  // lanes 0-15 carry K 0..7 (halves 0..7) + zeros (halves 8..15 -> K 16..23);
  // lanes 16-31 carry K 8..15 + zeros (K 24..31).
  v16bf a;
  #pragma unroll
  for (int j = 0; j < 16; ++j) a[j] = (__bf16)0.0f;
  {
    const float* arow = edge_attr + (size_t)(e0 + mrow) * EDGE_DIM + (hi ? 8 : 0);
    #pragma unroll
    for (int j = 0; j < 8; ++j) a[j] = (__bf16)arow[j];
  }

  // B: 32 x 16 (bf16). lanes 0-15 carry K 0..7 (halves 0..7), lanes 16-31
  // carry K 8..15; halves 8..15 (K 16..31) -> zero padding.
  v16bf b;
  #pragma unroll
  for (int j = 0; j < 16; ++j) {
    const float w = (j < 8) ? W_e[(size_t)((hi ? 8 : 0) + j) * D_OUT + ncol0 + n] : 0.0f;
    b[j] = (__bf16)w;
  }

  v8f Ec = {};
  Ec = __builtin_amdgcn_wmma_f32_16x16x32_bf16(
      false, a, false, b, (short)0, Ec, false, false);

  // Per-lane: 8 rows (edges) x this lane's column.  Gather x_l/x_r (L2
  // resident), LeakyReLU, scale by att, accumulate per-row partials.
  const int   mb   = hi ? 8 : 0;
  const float attv = att[ncol0 + n];
  float p[8];
  #pragma unroll
  for (int v = 0; v < 8; ++v) {
    const int m = mb + v;
    const int s = ei[e0 + m];
    const int d = ei[N_EDGES + e0 + m];
    float h = x_l[(size_t)s * D_OUT + ncol0 + n]
            + x_r[(size_t)d * D_OUT + ncol0 + n]
            + Ec[v];
    h = (h > 0.0f) ? h : NEG_SLOPE * h;
    p[v] = attv * h;
  }

  // Butterfly-reduce over the 16 lanes of each half (same row set per half).
  #pragma unroll
  for (int off = 1; off < 16; off <<= 1) {
    #pragma unroll
    for (int v = 0; v < 8; ++v) p[v] += __shfl_xor(p[v], off, 32);
  }

  // One lane per half folds this wave's 16-column contribution into LDS.
  if (n == 0) {
    #pragma unroll
    for (int v = 0; v < 8; ++v) atomicAdd(&sLogit[g * 16 + mb + v], p[v]);
  }
  __syncthreads();

  // Finalize the block's 32 edges: store logit, ordered-uint segment max.
  if (tid < 32) {
    const int   e     = blockIdx.x * 32 + tid;
    const int   dst   = ei[N_EDGES + e];
    const float logit = sLogit[tid];
    logits[e] = logit;
    unsigned u = __float_as_uint(logit);
    u = (u & 0x80000000u) ? ~u : (u | 0x80000000u);   // order-preserving encode
    atomicMax(&mEnc[dst], u);
  }
}

// ---------------------------------------------------------------------------
// Kernel 3: ex[e] = exp(logit[e] - m[dst]);  denom[dst] += ex[e]  (in place).
// ---------------------------------------------------------------------------
__global__ void edge_exp(const int* __restrict__ ei,
                         float* __restrict__ logits,
                         const unsigned* __restrict__ mEnc,
                         float* __restrict__ denom) {
  const int e = blockIdx.x * blockDim.x + threadIdx.x;
  if (e >= N_EDGES) return;
  const int dst = ei[N_EDGES + e];
  const unsigned u = mEnc[dst];
  const float m = __uint_as_float((u & 0x80000000u) ? (u ^ 0x80000000u) : ~u);
  const float ex = __expf(logits[e] - m);
  logits[e] = ex;
  atomicAdd(&denom[dst], ex);
}

// ---------------------------------------------------------------------------
// Kernel 4: out[dst] += (ex[e] / (denom[dst]+1e-16)) * x_l[src]
// 64 threads per edge: coalesced gather of x_l[src] and global_atomic_add_f32
// into out (both L2-resident).
// ---------------------------------------------------------------------------
__global__ void edge_scatter(const int* __restrict__ ei,
                             const float* __restrict__ exbuf,
                             const float* __restrict__ denom,
                             const float* __restrict__ x_l,
                             float* __restrict__ out) {
  const int t = blockIdx.x * blockDim.x + threadIdx.x;
  const int e = t >> 6;
  const int d = t & 63;
  if (e >= N_EDGES) return;
  const int src = ei[e];
  const int dst = ei[N_EDGES + e];
  const float alpha = exbuf[e] / (denom[dst] + 1e-16f);
  atomicAdd(&out[(size_t)dst * D_OUT + d], alpha * x_l[(size_t)src * D_OUT + d]);
}

// ---------------------------------------------------------------------------
extern "C" void kernel_launch(void* const* d_in, const int* in_sizes, int n_in,
                              void* d_out, int out_size, void* d_ws, size_t ws_size,
                              hipStream_t stream) {
  (void)in_sizes; (void)n_in; (void)out_size; (void)ws_size;

  const float* x         = (const float*)d_in[0];
  const int*   ei        = (const int*)d_in[1];   // harness convention: integer -> int32
  const float* edge_attr = (const float*)d_in[2];
  const float* W_l       = (const float*)d_in[3];
  const float* b_l       = (const float*)d_in[4];
  const float* W_r       = (const float*)d_in[5];
  const float* b_r       = (const float*)d_in[6];
  const float* W_e       = (const float*)d_in[7];
  const float* att       = (const float*)d_in[8];
  const float* bias      = (const float*)d_in[9];
  float* out = (float*)d_out;

  // Workspace carve-up (~30 MB total)
  char* ws = (char*)d_ws;
  float*    x_l    = (float*)ws;     ws += (size_t)N_NODES * D_OUT * sizeof(float);
  float*    x_r    = (float*)ws;     ws += (size_t)N_NODES * D_OUT * sizeof(float);
  float*    logits = (float*)ws;     ws += (size_t)N_EDGES * sizeof(float);
  unsigned* mEnc   = (unsigned*)ws;  ws += (size_t)N_NODES * sizeof(unsigned);
  float*    denom  = (float*)ws;

  gat_init<<<(N_NODES * D_OUT + 255) / 256, 256, 0, stream>>>(out, bias, mEnc, denom);
  node_transform_wmma<<<N_NODES / 16, 256, 0, stream>>>(x, W_l, b_l, W_r, b_r, x_l, x_r);
  edge_logits_wmma<<<N_EDGES / 32, 256, 0, stream>>>(x_l, x_r, ei, edge_attr,
                                                     W_e, att, logits, mEnc);
  edge_exp<<<(N_EDGES + 255) / 256, 256, 0, stream>>>(ei, logits, mEnc, denom);
  edge_scatter<<<(N_EDGES * 64 + 255) / 256, 256, 0, stream>>>(ei, logits, denom, x_l, out);
}
